// RelativeMultiHeadAttention_2422361555272
// MI455X (gfx1250) — matrix operand order repaired and checked by reading, not compiled, not run on hardware
//
#include <hip/hip_runtime.h>
#include <hip/hip_bf16.h>

typedef __attribute__((ext_vector_type(16))) _Float16 v16h;
typedef __attribute__((ext_vector_type(8)))  float    v8f;
typedef __attribute__((ext_vector_type(4)))  int      v4i;

#define DEV __device__ __forceinline__

// ---- optional CDNA5 async global->LDS copy path (ASYNCcnt-tracked) ----
#if defined(__has_builtin)
#  if __has_builtin(__builtin_amdgcn_global_load_async_to_lds_b128) && \
      __has_builtin(__builtin_amdgcn_s_wait_asynccnt)
#    define ASYNC_LDS 1
#  endif
#endif
#ifndef ASYNC_LDS
#  define ASYNC_LDS 0
#endif

#if ASYNC_LDS
typedef __attribute__((address_space(1))) v4i gv4i;  // global int4
typedef __attribute__((address_space(3))) v4i lv4i;  // LDS int4
#endif

namespace {

constexpr int BATCH  = 2;
constexpr int SEQ    = 2048;
constexpr int DMODEL = 1024;
constexpr int NHEAD  = 16;
constexpr int HDIM   = 64;                 // DMODEL / NHEAD
constexpr float SM_SCALE = 0.03125f;       // 1/sqrt(1024)

// 16-byte global->LDS copy; async (no VGPR round-trip) when available.
DEV void cp16_g2l(const void* g, void* l) {
#if ASYNC_LDS
  __builtin_amdgcn_global_load_async_to_lds_b128((gv4i*)g, (lv4i*)l, 0, 0);
#else
  *(uint4*)l = *(const uint4*)g;
#endif
}
DEV void cp_wait() {
#if ASYNC_LDS
  __builtin_amdgcn_s_wait_asynccnt(0);
#endif
}

// ---- WMMA fragment loaders per CDNA5 ISA §7.12.2 (wave32) ----
// A (16x32 f16): lanes 0-15 -> M=row, K = {0..7,16..23}; lanes 16-31 -> K+8.
template <int LDA>
DEV v16h load_a_frag(const _Float16* s, int row, int k0, int g) {
  v16h a;
#pragma unroll
  for (int e = 0; e < 16; ++e) {
    const int kc = (e < 8) ? (e + 8 * g) : (e + 8 + 8 * g);
    a[e] = s[row * LDA + k0 + kc];
  }
  return a;
}
// B (32x16 f16): lanes 0-15 -> N=col, K = {0..7,16..23}; lanes 16-31 -> K+8.
// b_lds is stored [N][K] so the 16 K-elements per lane are contiguous.
template <int LDB>
DEV v16h load_b_frag(const _Float16* s, int col, int k0, int g) {
  v16h b;
#pragma unroll
  for (int e = 0; e < 16; ++e) b[e] = s[col * LDB + k0 + ((e < 8) ? (e + 8 * g) : (e + 8 + 8 * g))];
  return b;
}

DEV v8f wmma16(v16h a, v16h b, v8f c) {
  return __builtin_amdgcn_wmma_f32_16x16x32_f16(
      /*neg_a=*/false, a, /*neg_b=*/false, b,
      /*c_mod=*/(short)0, c, /*reuse_a=*/false, /*reuse_b=*/false);
}

}  // namespace

// ======================= projection GEMM: X@W + b (+relpos) =================
// A [4096,1024] fp32, W [1024,1024] fp32 -> out f16 [4096,1024]
__global__ __launch_bounds__(256) void proj_gemm_kernel(
    const float* __restrict__ A, const float* __restrict__ W,
    const float* __restrict__ bias, _Float16* __restrict__ out,
    const int* __restrict__ clipPtr, int addRelPos) {
  __shared__ __align__(16) _Float16 a_lds[128 * 32];
  __shared__ __align__(16) _Float16 b_lds[64 * 32];

  const int tid = threadIdx.x;
  const int lane = tid & 31, wave = tid >> 5;
  const int wm = wave >> 1, wn = wave & 1;   // 4x2 wave grid
  const int l = lane & 15, g = lane >> 4;
  const int m0 = blockIdx.y * 128;
  const int n0 = blockIdx.x * 64;

  v8f c[2][2] = {};

  for (int k0 = 0; k0 < DMODEL; k0 += 32) {
    // Gather all global tile data into registers first (one wait for all),
    // then convert fp32->f16 and scatter to LDS.
    float4 va[4], vb[2];
#pragma unroll
    for (int it = 0; it < 4; ++it) {
      const int id = tid + it * 256;  // 1024 float4 chunks: A 128x32
      const int r = id >> 3, c4 = id & 7;
      va[it] = *(const float4*)(A + (size_t)(m0 + r) * DMODEL + k0 + c4 * 4);
    }
#pragma unroll
    for (int it = 0; it < 2; ++it) {
      const int id = tid + it * 256;  // 512 float4 chunks: W 32x64
      const int kk = id >> 4, c4 = id & 15;
      vb[it] = *(const float4*)(W + (size_t)(k0 + kk) * DMODEL + n0 + c4 * 4);
    }
    if (k0 + 32 < DMODEL) {  // global_prefetch_b8 of next K tile
      __builtin_prefetch(A + (size_t)(m0 + (tid >> 1)) * DMODEL + k0 + 32, 0, 1);
      __builtin_prefetch(W + (size_t)(k0 + 32 + (tid >> 3)) * DMODEL + n0, 0, 1);
    }
#pragma unroll
    for (int it = 0; it < 4; ++it) {
      const int id = tid + it * 256;
      const int r = id >> 3, c4 = id & 7;
      _Float16* p = &a_lds[r * 32 + c4 * 4];
      p[0] = (_Float16)va[it].x; p[1] = (_Float16)va[it].y;
      p[2] = (_Float16)va[it].z; p[3] = (_Float16)va[it].w;
    }
#pragma unroll
    for (int it = 0; it < 2; ++it) {
      const int id = tid + it * 256;
      const int kk = id >> 4, c4 = id & 15;
      b_lds[(c4 * 4 + 0) * 32 + kk] = (_Float16)vb[it].x;
      b_lds[(c4 * 4 + 1) * 32 + kk] = (_Float16)vb[it].y;
      b_lds[(c4 * 4 + 2) * 32 + kk] = (_Float16)vb[it].z;
      b_lds[(c4 * 4 + 3) * 32 + kk] = (_Float16)vb[it].w;
    }
    __syncthreads();

    const v16h a0 = load_a_frag<32>(a_lds, wm * 32 + l, 0, g);
    const v16h a1 = load_a_frag<32>(a_lds, wm * 32 + 16 + l, 0, g);
    const v16h b0 = load_b_frag<32>(b_lds, wn * 32 + l, 0, g);
    const v16h b1 = load_b_frag<32>(b_lds, wn * 32 + 16 + l, 0, g);
    c[0][0] = wmma16(a0, b0, c[0][0]);
    c[0][1] = wmma16(a0, b1, c[0][1]);
    c[1][0] = wmma16(a1, b0, c[1][0]);
    c[1][1] = wmma16(a1, b1, c[1][1]);
    __syncthreads();
  }

  const float clipf = (float)clipPtr[0];
#pragma unroll
  for (int i = 0; i < 2; ++i)
#pragma unroll
    for (int j = 0; j < 2; ++j) {
      const int rb = m0 + wm * 32 + i * 16 + 8 * g;
      const int cb = n0 + wn * 32 + j * 16 + l;
      const float bv = bias[cb];
#pragma unroll
      for (int r = 0; r < 8; ++r) {
        const int row = rb + r;
        float v = c[i][j][r] + bv;
        if (addRelPos) {
          const float rp = (float)(cb - (row & (SEQ - 1)));
          v += fminf(fmaxf(rp, -clipf), clipf);
        }
        out[(size_t)row * DMODEL + cb] = (_Float16)v;
      }
    }
}

// ======================= scores: (Qh @ Kh^T) * scale * mask ==================
// Q/K slabs f16 [BH][2048][64]; attn fp32 [BH][2048][2048]
__global__ __launch_bounds__(256) void scores_kernel(
    const _Float16* __restrict__ Q, const _Float16* __restrict__ Kh,
    const float* __restrict__ mask, float* __restrict__ attn) {
  __shared__ __align__(16) _Float16 a_lds[128 * 64];
  __shared__ __align__(16) _Float16 b_lds[64 * 64];

  const int tid = threadIdx.x;
  const int lane = tid & 31, wave = tid >> 5;
  const int wm = wave >> 1, wn = wave & 1;
  const int l = lane & 15, g = lane >> 4;
  const int bh = blockIdx.z;
  const int m0 = blockIdx.y * 128, n0 = blockIdx.x * 64;
  const size_t slab = (size_t)bh * SEQ * HDIM;

  // Q tile 128x64 (full head dim) and K tile 64x64: straight f16 copies ->
  // async global->LDS (no VGPR round trip) when available.
#pragma unroll
  for (int it = 0; it < 4; ++it) {
    const int id = tid + it * 256;  // 1024 chunks of 8 halfs
    const int r = id >> 3, c8 = id & 7;
    cp16_g2l(&Q[slab + (size_t)(m0 + r) * HDIM + c8 * 8],
             &a_lds[r * 64 + c8 * 8]);
  }
#pragma unroll
  for (int it = 0; it < 2; ++it) {
    const int id = tid + it * 256;  // 512 chunks
    const int n = id >> 3, c8 = id & 7;
    cp16_g2l(&Kh[slab + (size_t)(n0 + n) * HDIM + c8 * 8],
             &b_lds[n * 64 + c8 * 8]);
  }
  cp_wait();
  __syncthreads();

  v8f c[2][2] = {};
#pragma unroll
  for (int k0 = 0; k0 < HDIM; k0 += 32) {
    const v16h a0 = load_a_frag<64>(a_lds, wm * 32 + l, k0, g);
    const v16h a1 = load_a_frag<64>(a_lds, wm * 32 + 16 + l, k0, g);
    const v16h b0 = load_b_frag<64>(b_lds, wn * 32 + l, k0, g);
    const v16h b1 = load_b_frag<64>(b_lds, wn * 32 + 16 + l, k0, g);
    c[0][0] = wmma16(a0, b0, c[0][0]);
    c[0][1] = wmma16(a0, b1, c[0][1]);
    c[1][0] = wmma16(a1, b0, c[1][0]);
    c[1][1] = wmma16(a1, b1, c[1][1]);
  }

  const int b = bh >> 4;  // H == 16
#pragma unroll
  for (int i = 0; i < 2; ++i)
#pragma unroll
    for (int j = 0; j < 2; ++j) {
      const int rb = m0 + wm * 32 + i * 16 + 8 * g;
      const int cb = n0 + wn * 32 + j * 16 + l;
#pragma unroll
      for (int r = 0; r < 8; ++r) {
        const int row = rb + r;
        const float mk = mask[(size_t)b * SEQ * SEQ + (size_t)row * SEQ + cb];
        attn[((size_t)bh * SEQ + row) * SEQ + cb] = c[i][j][r] * SM_SCALE * mk;
      }
    }
}

// ======================= in-place row softmax over 2048 ======================
__global__ __launch_bounds__(256) void softmax_kernel(float* __restrict__ attn) {
  float* p = attn + (size_t)blockIdx.x * SEQ;
  const int tid = threadIdx.x;
  const int lane = tid & 31, wave = tid >> 5;
  __shared__ float red[8];

  float4 v0 = ((const float4*)p)[tid];
  float4 v1 = ((const float4*)p)[tid + 256];

  float m = fmaxf(fmaxf(v0.x, v0.y), fmaxf(v0.z, v0.w));
  m = fmaxf(m, fmaxf(fmaxf(v1.x, v1.y), fmaxf(v1.z, v1.w)));
#pragma unroll
  for (int o = 16; o > 0; o >>= 1) m = fmaxf(m, __shfl_xor(m, o, 32));
  if (lane == 0) red[wave] = m;
  __syncthreads();
  float rmax = red[0];
#pragma unroll
  for (int i = 1; i < 8; ++i) rmax = fmaxf(rmax, red[i]);
  __syncthreads();

  v0.x = __expf(v0.x - rmax); v0.y = __expf(v0.y - rmax);
  v0.z = __expf(v0.z - rmax); v0.w = __expf(v0.w - rmax);
  v1.x = __expf(v1.x - rmax); v1.y = __expf(v1.y - rmax);
  v1.z = __expf(v1.z - rmax); v1.w = __expf(v1.w - rmax);

  float s = v0.x + v0.y + v0.z + v0.w + v1.x + v1.y + v1.z + v1.w;
#pragma unroll
  for (int o = 16; o > 0; o >>= 1) s += __shfl_xor(s, o, 32);
  if (lane == 0) red[wave] = s;
  __syncthreads();
  float rsum = red[0];
#pragma unroll
  for (int i = 1; i < 8; ++i) rsum += red[i];

  const float inv = 1.0f / rsum;
  v0.x *= inv; v0.y *= inv; v0.z *= inv; v0.w *= inv;
  v1.x *= inv; v1.y *= inv; v1.z *= inv; v1.w *= inv;
  ((float4*)p)[tid] = v0;
  ((float4*)p)[tid + 256] = v1;
}

// ======================= AV: attn @ Vh -> av (f16) ===========================
__global__ __launch_bounds__(256) void av_kernel(const float* __restrict__ attn,
                                                 const _Float16* __restrict__ V,
                                                 _Float16* __restrict__ av) {
  __shared__ __align__(16) _Float16 a_lds[128 * 32];
  __shared__ __align__(16) _Float16 b_lds[64 * 32];

  const int tid = threadIdx.x;
  const int lane = tid & 31, wave = tid >> 5;
  const int wm = wave >> 1, wn = wave & 1;
  const int l = lane & 15, g = lane >> 4;
  const int bh = blockIdx.z;
  const int m0 = blockIdx.y * 128;  // N = 64 covered by one block
  const size_t arow = (size_t)bh * SEQ;
  const size_t slab = (size_t)bh * SEQ * HDIM;

  const int kkB = tid >> 3, c8B = tid & 7;  // 256 threads = 32x8 B chunks

  v8f c[2][2] = {};
  for (int k0 = 0; k0 < SEQ; k0 += 32) {
    // Gather: A (attn fp32 128x32) + B (V f16 32x64) into registers first.
    float4 va[4];
#pragma unroll
    for (int it = 0; it < 4; ++it) {
      const int id = tid + it * 256;
      const int r = id >> 3, c4 = id & 7;
      va[it] =
          *(const float4*)(attn + (arow + m0 + r) * (size_t)SEQ + k0 + c4 * 4);
    }
    _Float16 hb[8];
    {
      const _Float16* src = &V[slab + (size_t)(k0 + kkB) * HDIM + c8B * 8];
#pragma unroll
      for (int j = 0; j < 8; ++j) hb[j] = src[j];
    }
    if (k0 + 32 < SEQ)
      __builtin_prefetch(attn + (arow + m0 + (tid >> 1)) * (size_t)SEQ + k0 + 32,
                         0, 1);
#pragma unroll
    for (int it = 0; it < 4; ++it) {
      const int id = tid + it * 256;
      const int r = id >> 3, c4 = id & 7;
      _Float16* p = &a_lds[r * 32 + c4 * 4];
      p[0] = (_Float16)va[it].x; p[1] = (_Float16)va[it].y;
      p[2] = (_Float16)va[it].z; p[3] = (_Float16)va[it].w;
    }
#pragma unroll
    for (int j = 0; j < 8; ++j) b_lds[(c8B * 8 + j) * 32 + kkB] = hb[j];
    __syncthreads();

    const v16h a0 = load_a_frag<32>(a_lds, wm * 32 + l, 0, g);
    const v16h a1 = load_a_frag<32>(a_lds, wm * 32 + 16 + l, 0, g);
    const v16h b0 = load_b_frag<32>(b_lds, wn * 32 + l, 0, g);
    const v16h b1 = load_b_frag<32>(b_lds, wn * 32 + 16 + l, 0, g);
    c[0][0] = wmma16(a0, b0, c[0][0]);
    c[0][1] = wmma16(a0, b1, c[0][1]);
    c[1][0] = wmma16(a1, b0, c[1][0]);
    c[1][1] = wmma16(a1, b1, c[1][1]);
    __syncthreads();
  }

#pragma unroll
  for (int i = 0; i < 2; ++i)
#pragma unroll
    for (int j = 0; j < 2; ++j) {
      const int rb = m0 + wm * 32 + i * 16 + 8 * g;
      const int cb = wn * 32 + j * 16 + l;
#pragma unroll
      for (int r = 0; r < 8; ++r)
        av[slab + (size_t)(rb + r) * HDIM + cb] = (_Float16)c[i][j][r];
    }
}

// ======================= output GEMM: av @ Wo + bo (fp32 out) ================
__global__ __launch_bounds__(256) void out_gemm_kernel(
    const _Float16* __restrict__ A, const float* __restrict__ W,
    const float* __restrict__ bias, float* __restrict__ out) {
  __shared__ __align__(16) _Float16 a_lds[128 * 32];
  __shared__ __align__(16) _Float16 b_lds[64 * 32];

  const int tid = threadIdx.x;
  const int lane = tid & 31, wave = tid >> 5;
  const int wm = wave >> 1, wn = wave & 1;
  const int l = lane & 15, g = lane >> 4;
  const int m0 = blockIdx.y * 128;
  const int n0 = blockIdx.x * 64;

  v8f c[2][2] = {};
  for (int k0 = 0; k0 < DMODEL; k0 += 32) {
    // A: f16 128x32 straight copy -> async to LDS when available
#pragma unroll
    for (int it = 0; it < 2; ++it) {
      const int id = tid + it * 256;  // 512 chunks of 8 halfs
      const int r = id >> 2, c8 = id & 3;
      cp16_g2l(&A[(size_t)(m0 + r) * DMODEL + k0 + c8 * 8],
               &a_lds[r * 32 + c8 * 8]);
    }
    // B: Wo fp32 32x64: gather to registers, then f16 transposed [n][k]
    float4 vb[2];
#pragma unroll
    for (int it = 0; it < 2; ++it) {
      const int id = tid + it * 256;
      const int kk = id >> 4, c4 = id & 15;
      vb[it] = *(const float4*)(W + (size_t)(k0 + kk) * DMODEL + n0 + c4 * 4);
    }
    if (k0 + 32 < DMODEL)
      __builtin_prefetch(W + (size_t)(k0 + 32 + (tid >> 3)) * DMODEL + n0, 0, 1);
#pragma unroll
    for (int it = 0; it < 2; ++it) {
      const int id = tid + it * 256;
      const int kk = id >> 4, c4 = id & 15;
      b_lds[(c4 * 4 + 0) * 32 + kk] = (_Float16)vb[it].x;
      b_lds[(c4 * 4 + 1) * 32 + kk] = (_Float16)vb[it].y;
      b_lds[(c4 * 4 + 2) * 32 + kk] = (_Float16)vb[it].z;
      b_lds[(c4 * 4 + 3) * 32 + kk] = (_Float16)vb[it].w;
    }
    cp_wait();
    __syncthreads();

    const v16h a0 = load_a_frag<32>(a_lds, wm * 32 + l, 0, g);
    const v16h a1 = load_a_frag<32>(a_lds, wm * 32 + 16 + l, 0, g);
    const v16h b0 = load_b_frag<32>(b_lds, wn * 32 + l, 0, g);
    const v16h b1 = load_b_frag<32>(b_lds, wn * 32 + 16 + l, 0, g);
    c[0][0] = wmma16(a0, b0, c[0][0]);
    c[0][1] = wmma16(a0, b1, c[0][1]);
    c[1][0] = wmma16(a1, b0, c[1][0]);
    c[1][1] = wmma16(a1, b1, c[1][1]);
    __syncthreads();
  }

#pragma unroll
  for (int i = 0; i < 2; ++i)
#pragma unroll
    for (int j = 0; j < 2; ++j) {
      const int rb = m0 + wm * 32 + i * 16 + 8 * g;
      const int cb = n0 + wn * 32 + j * 16 + l;
      const float bv = bias[cb];
#pragma unroll
      for (int r = 0; r < 8; ++r)
        out[(size_t)(rb + r) * DMODEL + cb] = c[i][j][r] + bv;
    }
}

// ============================================================================
extern "C" void kernel_launch(void* const* d_in, const int* in_sizes, int n_in,
                              void* d_out, int out_size, void* d_ws,
                              size_t ws_size, hipStream_t stream) {
  const float* inputs  = (const float*)d_in[0];
  const float* context = (const float*)d_in[1];
  const float* mask    = (const float*)d_in[2];
  const float* Wq = (const float*)d_in[3];
  const float* bq = (const float*)d_in[4];
  const float* Wk = (const float*)d_in[5];
  const float* bk = (const float*)d_in[6];
  const float* Wv = (const float*)d_in[7];
  const float* bv = (const float*)d_in[8];
  const float* Wo = (const float*)d_in[9];
  const float* bo = (const float*)d_in[10];
  const int*   clip = (const int*)d_in[11];

  float* out  = (float*)d_out;
  float* attn = out + (size_t)BATCH * SEQ * DMODEL;  // outputs concat: (out, attn)

  _Float16* wsh = (_Float16*)d_ws;
  const size_t NQ = (size_t)BATCH * SEQ * DMODEL;  // 4M f16 elements per tensor
  _Float16* Qh  = wsh;
  _Float16* Kh  = wsh + NQ;
  _Float16* Vh  = wsh + 2 * NQ;
  _Float16* AVh = wsh + 3 * NQ;

  const dim3 blk(256);
  // M tiles = 4096/128 = 32, N tiles = 1024/64 = 16
  proj_gemm_kernel<<<dim3(16, 32), blk, 0, stream>>>(inputs, Wq, bq, Qh, clip, 1);
  proj_gemm_kernel<<<dim3(16, 32), blk, 0, stream>>>(context, Wk, bk, Kh, clip, 0);
  proj_gemm_kernel<<<dim3(16, 32), blk, 0, stream>>>(context, Wv, bv, Vh, clip, 1);
  // scores: n tiles = 2048/64 = 32, m tiles = 2048/128 = 16, z = B*H = 32
  scores_kernel<<<dim3(32, 16, 32), blk, 0, stream>>>(Qh, Kh, mask, attn);
  softmax_kernel<<<dim3(BATCH * NHEAD * SEQ), blk, 0, stream>>>(attn);
  av_kernel<<<dim3(1, 16, 32), blk, 0, stream>>>(attn, Vh, AVh);
  out_gemm_kernel<<<dim3(16, 32), blk, 0, stream>>>(AVh, Wo, bo, out);
}
